// STSA_66503273611804
// MI455X (gfx1250) — matrix operand order repaired and checked by reading, not compiled, not run on hardware
//
#include <hip/hip_runtime.h>

// Problem constants (reference: L,B,N,D,K = 32,4,1000,128,8; n_heads=8)
#define L_  32
#define B_  4
#define N_  1000
#define D_  128
#define K_  8
#define H_  8
#define DH_ 16

typedef __bf16 bf16_t;
typedef __attribute__((ext_vector_type(16))) __bf16 v16bf;
typedef __attribute__((ext_vector_type(8)))  __bf16 v8bf;
typedef __attribute__((ext_vector_type(4)))  __bf16 v4bf;
typedef __attribute__((ext_vector_type(8)))  float  v8f;

// ---------------------------------------------------------------------------
// WMMA helpers: D(16x16,f32) = A(16x32,bf16) * B(32x16,bf16) + C
// A fragment as loaded here (contiguous k, not ISA table order): lane&15 -> M row, half=lane>>4,
//   VGPR v, halfword j -> K = half*16 + 2v + j
// B fragment (same k order as A): lane&15 -> N col, K = half*16 + 2v + j
// C layout: M = half*8 + r (r = vgpr), N = lane&15
// ---------------------------------------------------------------------------
__device__ __forceinline__ v8f wmma_bf16(v16bf a, v16bf b, v8f c) {
  return __builtin_amdgcn_wmma_f32_16x16x32_bf16(
      /*neg_a=*/false, a, /*neg_b=*/false, b,
      /*c_mod=*/(short)0, c, /*reuse_a=*/false, /*reuse_b=*/false);
}

__device__ __forceinline__ v16bf ldA(const bf16_t* mat, int ld, int m0, int k0,
                                     int kvalid) {
  int lane = threadIdx.x & 31;
  int half = lane >> 4;
  const bf16_t* row = mat + (size_t)(m0 + (lane & 15)) * ld + k0;
  v16bf f;
#pragma unroll
  for (int v = 0; v < 8; ++v) {
#pragma unroll
    for (int j = 0; j < 2; ++j) {
      int k = half * 16 + v * 2 + j;
      f[v * 2 + j] = (k < kvalid) ? row[k] : (bf16_t)0.0f;
    }
  }
  return f;
}

// B fragment where B[k][n] = W[n0+n][k0+k] (W row-major, e.g. weight rows).
__device__ __forceinline__ v16bf ldB_rows(const bf16_t* W, int ld, int n0,
                                          int k0, int kvalid) {
  int lane = threadIdx.x & 31;
  int half = lane >> 4;
  const bf16_t* row = W + (size_t)(n0 + (lane & 15)) * ld + k0;
  v16bf f;
#pragma unroll
  for (int v = 0; v < 8; ++v) {
#pragma unroll
    for (int j = 0; j < 2; ++j) {
      int k = half * 16 + v * 2 + j;
      f[v * 2 + j] = (k < kvalid) ? row[k] : (bf16_t)0.0f;
    }
  }
  return f;
}

// B fragment where B[k][n] = mat[k0+k][c0+n] (mat row-major, rows = K).
__device__ __forceinline__ v16bf ldB_cols(const bf16_t* mat, int ld, int c0,
                                          int k0, int kvalid) {
  int lane = threadIdx.x & 31;
  int half = lane >> 4;
  int c = c0 + (lane & 15);
  v16bf f;
#pragma unroll
  for (int v = 0; v < 8; ++v) {
#pragma unroll
    for (int j = 0; j < 2; ++j) {
      int k = half * 16 + v * 2 + j;
      f[v * 2 + j] = (k < kvalid) ? mat[(size_t)(k0 + k) * ld + c] : (bf16_t)0.0f;
    }
  }
  return f;
}

// float4 -> packed 4x bf16 (8-byte LDS store)
__device__ __forceinline__ void st4(bf16_t* dst, float4 f) {
  v4bf h = {(bf16_t)f.x, (bf16_t)f.y, (bf16_t)f.z, (bf16_t)f.w};
  *(v4bf*)dst = h;
}

// ---------------------------------------------------------------------------
// Kernel 1: SK = x @ swk^T + sbk,  SV = x @ swv^T + sbv  (bf16 out, all T tok)
// ---------------------------------------------------------------------------
__global__ __launch_bounds__(256) void proj_kv_kernel(
    const float* __restrict__ x, const float* __restrict__ s_in_w,
    const float* __restrict__ s_in_b, bf16_t* __restrict__ SK,
    bf16_t* __restrict__ SV) {
  __shared__ bf16_t xt[64 * 136];
  __shared__ bf16_t wbuf[128 * 136];
  const int tid = threadIdx.x;
  const int t0 = blockIdx.x * 64;
  const int w = tid >> 5, lane = tid & 31, half = lane >> 4;

  for (int i = tid; i < 64 * (D_ / 4); i += 256) {
    int r = i >> 5, c = (i & 31) * 4;
    float4 f = *(const float4*)(x + (size_t)(t0 + r) * D_ + c);
    st4(xt + r * 136 + c, f);
  }

#pragma unroll 1
  for (int pass = 0; pass < 2; ++pass) {
    const float* W = s_in_w + (size_t)(pass + 1) * D_ * D_;  // swk then swv
    for (int i = tid; i < D_ * (D_ / 4); i += 256) {
      int r = i >> 5, c = (i & 31) * 4;
      float4 f = ((const float4*)W)[i];
      st4(wbuf + r * 136 + c, f);
    }
    __syncthreads();

    int mt = w >> 1;           // 0..3 (rows 64 = 4 tiles)
    int nbase = (w & 1) * 4;   // n-tiles nbase..nbase+3 (cols 128 = 8 tiles)
    v8f acc[4] = {{}, {}, {}, {}};
#pragma unroll
    for (int kc = 0; kc < 4; ++kc) {
      v16bf a = ldA(xt, 136, mt * 16, kc * 32, 32);
#pragma unroll
      for (int nt = 0; nt < 4; ++nt) {
        v16bf bb = ldB_rows(wbuf, 136, (nbase + nt) * 16, kc * 32, 32);
        acc[nt] = wmma_bf16(a, bb, acc[nt]);
      }
    }
    bf16_t* outp = (pass == 0) ? SK : SV;
    const float* bias = s_in_b + (pass + 1) * D_;
#pragma unroll
    for (int nt = 0; nt < 4; ++nt) {
      int col = (nbase + nt) * 16 + (lane & 15);
      float bv = bias[col];
#pragma unroll
      for (int r = 0; r < 8; ++r) {
        int row = t0 + mt * 16 + half * 8 + r;
        outp[(size_t)row * D_ + col] = (bf16_t)(acc[nt][r] + bv);
      }
    }
    __syncthreads();  // before wbuf is overwritten next pass
  }
}

// ---------------------------------------------------------------------------
// Kernel 2: temporal attention, one workgroup per (b,n).
// ---------------------------------------------------------------------------
__global__ __launch_bounds__(256) void temporal_kernel(
    const float* __restrict__ x, const float* __restrict__ t_in_w,
    const float* __restrict__ t_in_b, const float* __restrict__ t_out_w,
    const float* __restrict__ t_out_b, float* __restrict__ out) {
  __shared__ bf16_t xt[32 * 136];
  __shared__ bf16_t wbuf[128 * 136];
  __shared__ bf16_t qb[32 * 136];
  __shared__ bf16_t kb[32 * 136];
  __shared__ bf16_t vb[32 * 136];
  __shared__ float  sc[H_][32 * 36];
  __shared__ bf16_t pr[H_][32 * 36];
  __shared__ bf16_t ob[32 * 136];

  const int tid = threadIdx.x;
  const int n = blockIdx.x % N_;
  const int b = blockIdx.x / N_;
  const int w = tid >> 5, lane = tid & 31, half = lane >> 4;
  const float scale = 0.25f;  // 1/sqrt(dh=16)

  for (int i = tid; i < 32 * (D_ / 4); i += 256) {
    int l = i >> 5, c = (i & 31) * 4;
    float4 f = *(const float4*)(x + ((size_t)(l * B_ + b) * N_ + n) * D_ + c);
    st4(xt + l * 136 + c, f);
  }

  // q,k,v projections (weight buffer reused with barriers)
#pragma unroll 1
  for (int p = 0; p < 3; ++p) {
    __syncthreads();  // prior wbuf readers done (no-op cost at p==0)
    const float* W = t_in_w + (size_t)p * D_ * D_;
    for (int i = tid; i < D_ * (D_ / 4); i += 256) {
      int r = i >> 5, c = (i & 31) * 4;
      st4(wbuf + r * 136 + c, ((const float4*)W)[i]);
    }
    __syncthreads();
    // runtime-select LDS destination (pointer array of LDS addrs would become
    // an addrspacecast static initializer that lld rejects)
    bf16_t* dstp = (p == 0) ? qb : ((p == 1) ? kb : vb);
    int mt = w & 1;           // 32 rows = 2 m-tiles
    int n0 = (w >> 1) * 2;    // 128 cols = 8 n-tiles, 2 per wave
    v8f acc[2] = {{}, {}};
#pragma unroll
    for (int kc = 0; kc < 4; ++kc) {
      v16bf a = ldA(xt, 136, mt * 16, kc * 32, 32);
#pragma unroll
      for (int q = 0; q < 2; ++q) {
        v16bf bb = ldB_rows(wbuf, 136, (n0 + q) * 16, kc * 32, 32);
        acc[q] = wmma_bf16(a, bb, acc[q]);
      }
    }
#pragma unroll
    for (int q = 0; q < 2; ++q) {
      int col = (n0 + q) * 16 + (lane & 15);
      float bv = t_in_b[p * D_ + col];
#pragma unroll
      for (int r = 0; r < 8; ++r)
        dstp[(mt * 16 + half * 8 + r) * 136 + col] = (bf16_t)(acc[q][r] + bv);
    }
  }
  __syncthreads();

  // scores[h][l][m] = scale * q_h @ k_h^T   (K = dh = 16, zero-padded to 32)
  {
    int h = w;  // wave == head
    v8f acc[4] = {{}, {}, {}, {}};
#pragma unroll
    for (int mi = 0; mi < 2; ++mi) {
      v16bf a = ldA(qb, 136, mi * 16, h * DH_, 16);
#pragma unroll
      for (int ni = 0; ni < 2; ++ni) {
        v16bf bb = ldB_rows(kb, 136, ni * 16, h * DH_, 16);
        acc[mi * 2 + ni] = wmma_bf16(a, bb, acc[mi * 2 + ni]);
      }
    }
#pragma unroll
    for (int mi = 0; mi < 2; ++mi)
#pragma unroll
      for (int ni = 0; ni < 2; ++ni)
#pragma unroll
        for (int r = 0; r < 8; ++r)
          sc[h][(mi * 16 + half * 8 + r) * 36 + ni * 16 + (lane & 15)] =
              acc[mi * 2 + ni][r] * scale;
  }
  __syncthreads();

  // softmax over m: one thread per (h,l) -> 8*32 = 256
  {
    int h = tid >> 5, l = tid & 31;
    float mx = -1e30f;
    for (int m = 0; m < 32; ++m) mx = fmaxf(mx, sc[h][l * 36 + m]);
    float s = 0.0f;
    for (int m = 0; m < 32; ++m) s += __expf(sc[h][l * 36 + m] - mx);
    float inv = 1.0f / s;
    for (int m = 0; m < 32; ++m)
      pr[h][l * 36 + m] = (bf16_t)(__expf(sc[h][l * 36 + m] - mx) * inv);
  }
  __syncthreads();

  // o_h = P_h @ v_h  (K = 32 full)
  {
    int h = w;
    v16bf bb = ldB_cols(vb, 136, h * DH_, 0, 32);
#pragma unroll
    for (int mi = 0; mi < 2; ++mi) {
      v16bf a = ldA(pr[h], 36, mi * 16, 0, 32);
      v8f acc = {};
      acc = wmma_bf16(a, bb, acc);
#pragma unroll
      for (int r = 0; r < 8; ++r)
        ob[(mi * 16 + half * 8 + r) * 136 + h * DH_ + (lane & 15)] =
            (bf16_t)acc[r];
    }
  }
  __syncthreads();

  for (int i = tid; i < D_ * (D_ / 4); i += 256) {
    int r = i >> 5, c = (i & 31) * 4;
    st4(wbuf + r * 136 + c, ((const float4*)t_out_w)[i]);
  }
  __syncthreads();

  // out = ob @ t_out_w^T + t_out_b
  {
    int mt = w & 1;
    int n0 = (w >> 1) * 2;
    v8f acc[2] = {{}, {}};
#pragma unroll
    for (int kc = 0; kc < 4; ++kc) {
      v16bf a = ldA(ob, 136, mt * 16, kc * 32, 32);
#pragma unroll
      for (int q = 0; q < 2; ++q) {
        v16bf bb = ldB_rows(wbuf, 136, (n0 + q) * 16, kc * 32, 32);
        acc[q] = wmma_bf16(a, bb, acc[q]);
      }
    }
#pragma unroll
    for (int q = 0; q < 2; ++q) {
      int col = (n0 + q) * 16 + (lane & 15);
      float bv = t_out_b[col];
#pragma unroll
      for (int r = 0; r < 8; ++r) {
        int l = mt * 16 + half * 8 + r;
        out[((size_t)(l * B_ + b) * N_ + n) * D_ + col] = acc[q][r] + bv;
      }
    }
  }
}

// ---------------------------------------------------------------------------
// Kernel 3: spatial attention, one workgroup per (b,n); accumulates into out.
// ---------------------------------------------------------------------------
__global__ __launch_bounds__(256) void spatial_kernel(
    const float* __restrict__ x, const int* __restrict__ route_neighbors,
    const float* __restrict__ s_in_w, const float* __restrict__ s_in_b,
    const float* __restrict__ s_out_w, const float* __restrict__ s_out_b,
    const bf16_t* __restrict__ SK, const bf16_t* __restrict__ SV,
    float* __restrict__ out) {
  __shared__ bf16_t xt[32 * 136];
  __shared__ bf16_t wbuf[128 * 136];
  __shared__ float  sq[32 * 132];
  __shared__ bf16_t so[32 * 136];
  __shared__ int    nbr[K_];

  const int tid = threadIdx.x;
  const int n = blockIdx.x % N_;
  const int b = blockIdx.x / N_;
  const int w = tid >> 5, lane = tid & 31, half = lane >> 4;
  const float scale = 0.25f;

  if (tid < K_) nbr[tid] = route_neighbors[n * K_ + tid];
  for (int i = tid; i < 32 * (D_ / 4); i += 256) {
    int l = i >> 5, c = (i & 31) * 4;
    float4 f = *(const float4*)(x + ((size_t)(l * B_ + b) * N_ + n) * D_ + c);
    st4(xt + l * 136 + c, f);
  }
  for (int i = tid; i < D_ * (D_ / 4); i += 256) {
    int r = i >> 5, c = (i & 31) * 4;
    st4(wbuf + r * 136 + c, ((const float4*)s_in_w)[i]);  // swq
  }
  __syncthreads();

  // sq = x @ swq^T + sbq  (f32 in LDS)
  {
    int mt = w & 1;
    int n0 = (w >> 1) * 2;
    v8f acc[2] = {{}, {}};
#pragma unroll
    for (int kc = 0; kc < 4; ++kc) {
      v16bf a = ldA(xt, 136, mt * 16, kc * 32, 32);
#pragma unroll
      for (int q = 0; q < 2; ++q) {
        v16bf bb = ldB_rows(wbuf, 136, (n0 + q) * 16, kc * 32, 32);
        acc[q] = wmma_bf16(a, bb, acc[q]);
      }
    }
#pragma unroll
    for (int q = 0; q < 2; ++q) {
      int col = (n0 + q) * 16 + (lane & 15);
      float bv = s_in_b[col];
#pragma unroll
      for (int r = 0; r < 8; ++r)
        sq[(mt * 16 + half * 8 + r) * 132 + col] = acc[q][r] + bv;
    }
  }
  __syncthreads();

  // K=8 neighbor attention: one thread per (l,h); bf16 gather of SK/SV
  {
    int h = tid & 7, l = tid >> 3;
    const float* qrow = sq + l * 132 + h * DH_;
    float qv[DH_];
#pragma unroll
    for (int jj = 0; jj < 4; ++jj) {
      float4 f = *(const float4*)(qrow + jj * 4);
      qv[jj * 4 + 0] = f.x; qv[jj * 4 + 1] = f.y;
      qv[jj * 4 + 2] = f.z; qv[jj * 4 + 3] = f.w;
    }
    size_t base_l = (size_t)(l * B_ + b) * N_;
    float scv[K_];
    float mx = -1e30f;
#pragma unroll
    for (int k = 0; k < K_; ++k) {
      size_t g = (base_l + nbr[k]) * D_ + h * DH_;
      const v8bf* kp = (const v8bf*)(SK + g);   // 16B-aligned: g % 16 == 0
      v8bf k0 = kp[0], k1 = kp[1];
      float d = 0.0f;
#pragma unroll
      for (int j = 0; j < 8; ++j) {
        d += qv[j] * (float)k0[j];
        d += qv[8 + j] * (float)k1[j];
      }
      scv[k] = d * scale;
      mx = fmaxf(mx, scv[k]);
    }
    float s = 0.0f;
#pragma unroll
    for (int k = 0; k < K_; ++k) { scv[k] = __expf(scv[k] - mx); s += scv[k]; }
    float inv = 1.0f / s;
    float acc[DH_];
#pragma unroll
    for (int j = 0; j < DH_; ++j) acc[j] = 0.0f;
#pragma unroll
    for (int k = 0; k < K_; ++k) {
      float p = scv[k] * inv;
      size_t g = (base_l + nbr[k]) * D_ + h * DH_;
      const v8bf* vp = (const v8bf*)(SV + g);
      v8bf v0 = vp[0], v1 = vp[1];
#pragma unroll
      for (int j = 0; j < 8; ++j) {
        acc[j]     += p * (float)v0[j];
        acc[8 + j] += p * (float)v1[j];
      }
    }
    bf16_t* sor = so + l * 136 + h * DH_;
#pragma unroll
    for (int jj = 0; jj < 4; ++jj) {
      v4bf hh = {(bf16_t)acc[jj * 4 + 0], (bf16_t)acc[jj * 4 + 1],
                 (bf16_t)acc[jj * 4 + 2], (bf16_t)acc[jj * 4 + 3]};
      *(v4bf*)(sor + jj * 4) = hh;
    }
  }
  __syncthreads();

  for (int i = tid; i < D_ * (D_ / 4); i += 256) {
    int r = i >> 5, c = (i & 31) * 4;
    st4(wbuf + r * 136 + c, ((const float4*)s_out_w)[i]);
  }
  __syncthreads();

  // out += so @ s_out_w^T + s_out_b
  {
    int mt = w & 1;
    int n0 = (w >> 1) * 2;
    v8f acc[2] = {{}, {}};
#pragma unroll
    for (int kc = 0; kc < 4; ++kc) {
      v16bf a = ldA(so, 136, mt * 16, kc * 32, 32);
#pragma unroll
      for (int q = 0; q < 2; ++q) {
        v16bf bb = ldB_rows(wbuf, 136, (n0 + q) * 16, kc * 32, 32);
        acc[q] = wmma_bf16(a, bb, acc[q]);
      }
    }
#pragma unroll
    for (int q = 0; q < 2; ++q) {
      int col = (n0 + q) * 16 + (lane & 15);
      float bv = s_out_b[col];
#pragma unroll
      for (int r = 0; r < 8; ++r) {
        int l = mt * 16 + half * 8 + r;
        size_t g = ((size_t)(l * B_ + b) * N_ + n) * D_ + col;
        out[g] = out[g] + acc[q][r] + bv;
      }
    }
  }
}

// ---------------------------------------------------------------------------
extern "C" void kernel_launch(void* const* d_in, const int* in_sizes, int n_in,
                              void* d_out, int out_size, void* d_ws,
                              size_t ws_size, hipStream_t stream) {
  (void)in_sizes; (void)n_in; (void)out_size; (void)ws_size;
  const float* x        = (const float*)d_in[0];
  const int*   route    = (const int*)d_in[1];
  const float* t_in_w   = (const float*)d_in[2];
  const float* t_in_b   = (const float*)d_in[3];
  const float* t_out_w  = (const float*)d_in[4];
  const float* t_out_b  = (const float*)d_in[5];
  const float* s_in_w   = (const float*)d_in[6];
  const float* s_in_b   = (const float*)d_in[7];
  const float* s_out_w  = (const float*)d_in[8];
  const float* s_out_b  = (const float*)d_in[9];
  float* out = (float*)d_out;

  const size_t T = (size_t)L_ * B_ * N_;   // 128000 tokens
  bf16_t* SK = (bf16_t*)d_ws;              // T * 128 bf16
  bf16_t* SV = SK + T * D_;                // T * 128 bf16  (65.5 MB total)

  proj_kv_kernel<<<(int)(T / 64), 256, 0, stream>>>(x, s_in_w, s_in_b, SK, SV);
  temporal_kernel<<<B_ * N_, 256, 0, stream>>>(x, t_in_w, t_in_b, t_out_w,
                                               t_out_b, out);
  spatial_kernel<<<B_ * N_, 256, 0, stream>>>(x, route, s_in_w, s_in_b,
                                              s_out_w, s_out_b, SK, SV, out);
}
